// ContLossforCluster_ALL_3478923510128
// MI455X (gfx1250) — matrix operand order repaired and checked by reading, not compiled, not run on hardware
//
#include <hip/hip_runtime.h>
#include <hip/hip_bf16.h>

// ---------------------------------------------------------------------------
// Fused supervised-contrastive loss for MI455X (gfx1250, wave32, WMMA).
//   sim = (f_hat @ g_hatT) / T via v_wmma_f32_16x16x32_f16, consumed
//   in-register/LDS (never materialized: saves ~1 GB of HBM round-trip).
// Round-2 improvements: M=64 row blocks (4x B-register reuse, 4 independent
// WMMA accumulator chains), software-pipelined B (ping-pong register
// buffers), double-buffered LDS tile (one barrier/iteration), int4 cluster
// loads, float4 LDS scans.
// ---------------------------------------------------------------------------

typedef _Float16 h16 __attribute__((ext_vector_type(16)));
typedef _Float16 h8  __attribute__((ext_vector_type(8)));
typedef _Float16 h4  __attribute__((ext_vector_type(4)));
typedef float    f8  __attribute__((ext_vector_type(8)));

#define B_ROWS      2048
#define N_COLS      65536
#define D_DIM       128
#define MBLK        64                    // rows per workgroup (4 WMMA row-sets)
#define NSPLIT      32                    // N-dimension partitions
#define COLS_SPLIT  (N_COLS / NSPLIT)     // 2048
#define ITERS       (COLS_SPLIT / 128)    // 16 iterations of 64x128
#define KNN         10
#define INV_T       (1.0f / 0.07f)
#define EPSF        1e-12f
#define NEG_INF     (-3.0e38f)

// ---------------------------------------------------------------------------
// Row L2-normalize (D=128) + f32 -> f16.  One wave32 per row.
// ---------------------------------------------------------------------------
__global__ __launch_bounds__(256) void normalize_rows_f16(
    const float* __restrict__ in, _Float16* __restrict__ out, int rows) {
  int gid  = blockIdx.x * blockDim.x + threadIdx.x;
  int wave = gid >> 5;
  int lane = threadIdx.x & 31;
  if (wave >= rows) return;
  const float4 v = ((const float4*)(in + (size_t)wave * D_DIM))[lane];
  float ss = v.x * v.x + v.y * v.y + v.z * v.z + v.w * v.w;
  #pragma unroll
  for (int off = 16; off > 0; off >>= 1) ss += __shfl_xor(ss, off, 32);
  float scale = 1.0f / fmaxf(sqrtf(ss), EPSF);
  h4 o;
  o[0] = (_Float16)(v.x * scale);
  o[1] = (_Float16)(v.y * scale);
  o[2] = (_Float16)(v.z * scale);
  o[3] = (_Float16)(v.w * scale);
  ((h4*)(out + (size_t)wave * D_DIM))[lane] = o;
}

__global__ void init_accum(float* accum) { accum[0] = 0.0f; }

// ---------------------------------------------------------------------------
// Fused GEMM + row statistics.
//   grid = (B_ROWS/MBLK) * NSPLIT blocks, 256 threads (8 waves).
//   Per iteration each wave computes a 64x16 strip (4 row-sets x 16 cols,
//   16 WMMAs) reusing one register-resident B panel 4x; the scaled 64x128
//   score tile lands in a double-buffered LDS tile consumed by a per-thread
//   (row, 32-col) scan: sum(exp), cluster positives, sorted top-10.
// ---------------------------------------------------------------------------
__global__ __launch_bounds__(256) void fused_sim_stats(
    const _Float16* __restrict__ fh, const _Float16* __restrict__ gh,
    const int* __restrict__ fclus, const int* __restrict__ gclus,
    float* __restrict__ p_sumexp, float* __restrict__ p_possum,
    int* __restrict__ p_cnt, float* __restrict__ p_topv,
    int* __restrict__ p_topi) {
  const int tid  = threadIdx.x;
  const int wave = tid >> 5;
  const int lane = tid & 31;
  const int m    = lane & 15;   // A row (within set) / B column within tile
  const int hi   = lane >> 4;   // lane half (ISA 16-bit operand layout)

  const int rowblk = blockIdx.x >> 5;   // / NSPLIT
  const int split  = blockIdx.x & 31;   // % NSPLIT
  const int row0   = rowblk * MBLK;
  const int col0   = split * COLS_SPLIT;

  __shared__ float tile[2][MBLK][132];  // double-buffered 64x128 scores (+pad)
  __shared__ float s_se[MBLK];          // per-row sum(exp)
  __shared__ float s_ps[MBLK];          // per-row positive-sim sum
  __shared__ int   s_pc[MBLK];          // per-row positive count
  __shared__ int   s_mc[MBLK];          // per-row cluster id
  __shared__ float s_mval[MBLK][40];    // 4 sub-lists x 10 per row
  __shared__ int   s_midx[MBLK][40];

  if (tid < MBLK) {
    s_se[tid] = 0.0f;
    s_ps[tid] = 0.0f;
    s_pc[tid] = 0;
    s_mc[tid] = fclus[row0 + tid];
  }
  __syncthreads();

  // ---- A (64x128 f16) register-resident: 4 row-sets x 4 k-steps.
  // A layout per k32 (matches B): lane half `hi` holds K = hi*16+j
  // (j = 0..15) of row m.
  union AU { h16 v; h8 h[2]; };
  AU A[4][4];
  #pragma unroll
  for (int s = 0; s < 4; ++s) {
    const _Float16* frow = fh + (size_t)(row0 + s * 16 + m) * D_DIM;
    #pragma unroll
    for (int kk = 0; kk < 4; ++kk) {
      A[s][kk].h[0] = *(const h8*)(frow + kk * 32 + hi * 16);
      A[s][kk].h[1] = *(const h8*)(frow + kk * 32 + hi * 16 + 8);
    }
  }

  // ---- per-thread scan state: row r = tid>>2, columns sub*32 .. +31
  const int r   = tid >> 2;
  const int sub = tid & 3;
  const int myclus = s_mc[r];
  float sumexp = 0.0f, possum = 0.0f;
  int   cnt = 0;
  float topv[10];
  int   topi[10];
  #pragma unroll
  for (int q = 0; q < 10; ++q) { topv[q] = NEG_INF; topi[q] = -1; }

  // B panel loader: per lane one contiguous 32B chunk per k-step
  // (B layout: lane half `hi` holds K = hi*16 + j of g-row/column m).
  auto loadB = [&](h16* Bt, int itx) {
    const _Float16* gr =
        gh + (size_t)(col0 + itx * 128 + wave * 16 + m) * D_DIM + hi * 16;
    #pragma unroll
    for (int kk = 0; kk < 4; ++kk) Bt[kk] = *(const h16*)(gr + kk * 32);
  };

  // One pipelined iteration: prefetch B for `itpre` into Bn, compute with Bc,
  // store into tiles[buf], barrier, scan tiles[buf].
  auto body = [&](const h16* Bc, h16* Bn, int it, int itpre, int buf) {
    loadB(Bn, itpre);

    f8 z = {};
    f8 c[4] = {z, z, z, z};
    #pragma unroll
    for (int kk = 0; kk < 4; ++kk) {
      #pragma unroll
      for (int s = 0; s < 4; ++s)   // 4 independent accumulator chains
        c[s] = __builtin_amdgcn_wmma_f32_16x16x32_f16(
            false, A[s][kk].v, false, Bc[kk], (short)0, c[s], false, false);
    }
    // C/D layout: VGPR i, lane -> (M = s*16 + hi*8 + i, N = m)
    #pragma unroll
    for (int s = 0; s < 4; ++s)
      #pragma unroll
      for (int i = 0; i < 8; ++i)
        tile[buf][s * 16 + hi * 8 + i][wave * 16 + m] = c[s][i] * INV_T;
    __syncthreads();

    // ---- consume: exp-sum, cluster positives, top-10 over 32 columns
    const int colg0 = col0 + it * 128 + sub * 32;
    #pragma unroll
    for (int jj = 0; jj < 8; ++jj) {
      const float4 tv = *(const float4*)&tile[buf][r][sub * 32 + jj * 4];
      const int4   gc = *(const int4*)(gclus + colg0 + jj * 4);
      const float  sv[4] = {tv.x, tv.y, tv.z, tv.w};
      const int    gv[4] = {gc.x, gc.y, gc.z, gc.w};
      #pragma unroll
      for (int j2 = 0; j2 < 4; ++j2) {
        const float s = sv[j2];
        sumexp += __expf(s);
        if (gv[j2] == myclus) { possum += s; ++cnt; }
        if (s > topv[9]) {                  // guarded sorted-insert
          float v = s; int id = colg0 + jj * 4 + j2;
          #pragma unroll
          for (int q = 0; q < 10; ++q) {
            if (v > topv[q]) {
              float t0 = topv[q]; int t1 = topi[q];
              topv[q] = v; topi[q] = id;
              v = t0; id = t1;
            }
          }
        }
      }
    }
    // no trailing barrier: next iteration writes the other LDS buffer, and
    // its barrier orders this scan against the buffer's next overwrite.
  };

  h16 B0[4], B1[4];
  loadB(B0, 0);
  for (int itp = 0; itp < ITERS; itp += 2) {
    const int pre2 = (itp + 2 < ITERS) ? itp + 2 : ITERS - 1;
    body(B0, B1, itp, itp + 1, 0);
    body(B1, B0, itp + 1, pre2, 1);
  }

  // ---- combine the 4 sub-scans of each row
  atomicAdd(&s_se[r], sumexp);
  atomicAdd(&s_ps[r], possum);
  atomicAdd(&s_pc[r], cnt);
  #pragma unroll
  for (int q = 0; q < 10; ++q) {
    s_mval[r][sub * 10 + q] = topv[q];
    s_midx[r][sub * 10 + q] = topi[q];
  }
  __syncthreads();

  if (tid < MBLK) {  // one thread per row: merge 40 candidates -> top-10
    float mv[10]; int mi[10];
    #pragma unroll
    for (int q = 0; q < 10; ++q) { mv[q] = NEG_INF; mi[q] = -1; }
    for (int j = 0; j < 40; ++j) {
      float v = s_mval[tid][j];
      if (v > mv[9]) {
        int id = s_midx[tid][j];
        #pragma unroll
        for (int q = 0; q < 10; ++q) {
          if (v > mv[q]) {
            float t0 = mv[q]; int t1 = mi[q];
            mv[q] = v; mi[q] = id;
            v = t0; id = t1;
          }
        }
      }
    }
    const size_t pidx = (size_t)(row0 + tid) * NSPLIT + split;
    #pragma unroll
    for (int q = 0; q < 10; ++q) {
      p_topv[pidx * 10 + q] = mv[q];
      p_topi[pidx * 10 + q] = mi[q];
    }
    p_sumexp[pidx] = s_se[tid];
    p_possum[pidx] = s_ps[tid];
    p_cnt[pidx]    = s_pc[tid];
  }
}

// ---------------------------------------------------------------------------
// Merge the NSPLIT partials of each row: global top-10 via guarded sorted
// insert over 320 candidates, union with the cluster mask, reduce.
// ---------------------------------------------------------------------------
__global__ __launch_bounds__(256) void merge_rows(
    const float* __restrict__ p_sumexp, const float* __restrict__ p_possum,
    const int* __restrict__ p_cnt, const float* __restrict__ p_topv,
    const int* __restrict__ p_topi, const int* __restrict__ fclus,
    const int* __restrict__ gclus, float* __restrict__ accum) {
  const int row = blockIdx.x * blockDim.x + threadIdx.x;
  if (row >= B_ROWS) return;

  float sumexp = 0.0f, possum = 0.0f;
  int cnt = 0;
  #pragma unroll 8
  for (int sp = 0; sp < NSPLIT; ++sp) {
    const size_t p = (size_t)row * NSPLIT + sp;
    sumexp += p_sumexp[p];
    possum += p_possum[p];
    cnt    += p_cnt[p];
  }

  // global top-10 of the 320 per-split candidates
  float tv[10]; int ti[10];
  #pragma unroll
  for (int q = 0; q < 10; ++q) { tv[q] = NEG_INF; ti[q] = -1; }
  const size_t cand0 = (size_t)row * NSPLIT * 10;
  for (int j = 0; j < NSPLIT * 10; ++j) {
    float v = p_topv[cand0 + j];
    if (v > tv[9]) {
      int id = p_topi[cand0 + j];
      #pragma unroll
      for (int q = 0; q < 10; ++q) {
        if (v > tv[q]) {
          float t0 = tv[q]; int t1 = ti[q];
          tv[q] = v; ti[q] = id;
          v = t0; id = t1;
        }
      }
    }
  }

  const int myc = fclus[row];
  #pragma unroll
  for (int q = 0; q < KNN; ++q) {       // union with cluster mask
    if (gclus[ti[q]] != myc) { possum += tv[q]; ++cnt; }
  }

  const float mlpp =
      (possum - (float)cnt * __logf(sumexp + EPSF)) / ((float)cnt + EPSF);
  atomicAdd(accum, mlpp);
}

__global__ void finalize_loss(const float* __restrict__ accum,
                              float* __restrict__ out) {
  out[0] = -accum[0] / (float)B_ROWS;
}

// ---------------------------------------------------------------------------
extern "C" void kernel_launch(void* const* d_in, const int* in_sizes, int n_in,
                              void* d_out, int out_size, void* d_ws,
                              size_t ws_size, hipStream_t stream) {
  (void)in_sizes; (void)n_in; (void)out_size; (void)ws_size;

  const float* features        = (const float*)d_in[0];
  const int*   cluster_idxes   = (const int*)d_in[1];
  const float* global_features = (const float*)d_in[2];
  const int*   global_clusters = (const int*)d_in[3];
  float*       out             = (float*)d_out;

  // --- workspace layout (all segments 256B aligned)
  char* ws = (char*)d_ws;
  size_t off = 0;
  float* accum    = (float*)(ws + off);  off += 256;
  float* p_sumexp = (float*)(ws + off);  off += (size_t)B_ROWS * NSPLIT * 4;
  float* p_possum = (float*)(ws + off);  off += (size_t)B_ROWS * NSPLIT * 4;
  int*   p_cnt    = (int*)(ws + off);    off += (size_t)B_ROWS * NSPLIT * 4;
  float* p_topv   = (float*)(ws + off);  off += (size_t)B_ROWS * NSPLIT * 10 * 4;
  int*   p_topi   = (int*)(ws + off);    off += (size_t)B_ROWS * NSPLIT * 10 * 4;
  _Float16* fh    = (_Float16*)(ws + off); off += (size_t)B_ROWS * D_DIM * 2;
  _Float16* gh    = (_Float16*)(ws + off); off += (size_t)N_COLS * D_DIM * 2;

  init_accum<<<1, 1, 0, stream>>>(accum);

  normalize_rows_f16<<<B_ROWS / 8, 256, 0, stream>>>(features, fh, B_ROWS);
  normalize_rows_f16<<<N_COLS / 8, 256, 0, stream>>>(global_features, gh, N_COLS);

  fused_sim_stats<<<(B_ROWS / MBLK) * NSPLIT, 256, 0, stream>>>(
      fh, gh, cluster_idxes, global_clusters,
      p_sumexp, p_possum, p_cnt, p_topv, p_topi);

  merge_rows<<<(B_ROWS + 255) / 256, 256, 0, stream>>>(
      p_sumexp, p_possum, p_cnt, p_topv, p_topi,
      cluster_idxes, global_clusters, accum);

  finalize_loss<<<1, 1, 0, stream>>>(accum, out);
}
